// DeepND_ST_61572651156107
// MI455X (gfx1250) — matrix operand order repaired and checked by reading, not compiled, not run on hardware
//
#include <hip/hip_runtime.h>
#include <hip/hip_bf16.h>

#define N_GENES 100000
#define E_EDGES 1600000
#define NEXPD   16
#define UNITD   15
#define H1D     64
#define FEATD   20
#define EPSV    1e-5f

typedef __attribute__((ext_vector_type(16))) _Float16 v16h;
typedef __attribute__((ext_vector_type(8)))  _Float16 v8h;
typedef __attribute__((ext_vector_type(8)))  float    v8f;

// ---------------- utility ----------------
__global__ void k_zero(float* __restrict__ p, int n) {
  int i = blockIdx.x * blockDim.x + threadIdx.x;
  if (i < n) p[i] = 0.f;
}

// gating: softmax(gf @ Wg + bg) over 16 experts, per node
__global__ void k_gate(const float* __restrict__ gf, const float* __restrict__ Wg,
                       const float* __restrict__ bg, float* __restrict__ wgate) {
  int n = blockIdx.x * blockDim.x + threadIdx.x;
  if (n >= N_GENES) return;
  float l[NEXPD];
#pragma unroll
  for (int i = 0; i < NEXPD; ++i) l[i] = bg[i];
  const float* g = gf + n * FEATD;
  for (int f = 0; f < FEATD; ++f) {
    float gv = g[f];
#pragma unroll
    for (int i = 0; i < NEXPD; ++i) l[i] += gv * Wg[f * NEXPD + i];
  }
  float m = l[0];
#pragma unroll
  for (int i = 1; i < NEXPD; ++i) m = fmaxf(m, l[i]);
  float s = 0.f;
#pragma unroll
  for (int i = 0; i < NEXPD; ++i) { l[i] = expf(l[i] - m); s += l[i]; }
  float inv = 1.f / s;
#pragma unroll
  for (int i = 0; i < NEXPD; ++i) wgate[n * NEXPD + i] = l[i] * inv;
}

// convert x [N,15] -> f16 padded [N,16] (done once, shared by all experts)
__global__ void k_xconv(const float* __restrict__ x, _Float16* __restrict__ xh) {
  int idx = blockIdx.x * blockDim.x + threadIdx.x;   // exactly N*16 threads
  int n = idx >> 4, k = idx & 15;
  xh[idx] = (k < UNITD) ? (_Float16)x[n * UNITD + k] : (_Float16)0.f;
}

// Pre-swizzle W1[i] [15,64] into WMMA B fragments, f16.
// Layout: Bfrag[(ct*32 + lane)*16 + j] = W1[k, ct*16 + (lane&15)], k = j<8 ? kb+j : 16+kb+(j-8)
// with kb = (lane>>4)*8, zero-padded past K=15.  Each lane of k_gemm1 then
// loads its whole v16h fragment as one contiguous 32B vector load.
__global__ void k_bswz(const float* __restrict__ W1i, _Float16* __restrict__ Bfrag) {
  int idx = blockIdx.x * blockDim.x + threadIdx.x;   // exactly 4*32*16 = 2048 threads
  int ct   = idx >> 9;          // 0..3  column tile
  int lane = (idx >> 4) & 31;   // 0..31
  int j    = idx & 15;          // 0..15 element within fragment
  int m  = lane & 15;
  int kb = (lane >> 4) << 3;    // 0 or 8
  int k  = (j < 8) ? kb + j : 16 + kb + (j - 8);
  int col = (ct << 4) + m;
  Bfrag[idx] = (k < UNITD) ? (_Float16)W1i[k * 64 + col] : (_Float16)0.f;
}

// in-degree via f32 atomics (exact: counts << 2^24)
__global__ void k_deg(const int* __restrict__ dst, float* __restrict__ deg) {
  int e = blockIdx.x * blockDim.x + threadIdx.x;     // exactly E threads
  atomicAdd(&deg[dst[e]], 1.0f);
}

// deg -> dinv = rsqrt(deg + 2) in place (improved=True adds 2I)
__global__ void k_dinv(float* __restrict__ degv) {
  int n = blockIdx.x * blockDim.x + threadIdx.x;
  if (n < N_GENES) degv[n] = rsqrtf(degv[n] + 2.0f);
}

// h1 = x @ W1 via WMMA f16->f32. One 16x16x32 WMMA covers all of K=15.
// 2 waves/block, one 16-row x 64-col tile per wave, 4 WMMAs per tile.
// A: single 16B vector load of the 8 live halves; B: single 32B vector load
// from the pre-swizzled fragment buffer (identical for all tiles, L0-resident).
__global__ void k_gemm1(const _Float16* __restrict__ xh, const _Float16* __restrict__ Bfrag,
                        float* __restrict__ h1) {
  int lane = threadIdx.x & 31;
  int wave = threadIdx.x >> 5;
  int tile = blockIdx.x * 2 + wave;                  // 0..6249 exact (N = 6250*16)
  int row0 = tile << 4;
  int m  = lane & 15;
  int hi = lane >> 4;
  // A fragment: 16-bit A 16x32 layout. Lane-half hi: VGPR v in 0..3 holds
  // K = hi*8 + 2v, 2v+1 ; VGPRs 4..7 hold K=16..31 (all zero padding).
  v8h alo = *(const v8h*)(xh + (row0 + m) * 16 + hi * 8);   // 16B aligned
  v16h a;
#pragma unroll
  for (int j = 0; j < 8; ++j) a[j] = alo[j];
#pragma unroll
  for (int j = 8; j < 16; ++j) a[j] = (_Float16)0.f;
  const v16h* bf = (const v16h*)Bfrag;                       // 32B-aligned fragments
#pragma unroll
  for (int ct = 0; ct < 4; ++ct) {
    v16h b = bf[(ct << 5) + lane];
    v8f c = {0.f, 0.f, 0.f, 0.f, 0.f, 0.f, 0.f, 0.f};
    c = __builtin_amdgcn_wmma_f32_16x16x32_f16(false, a, false, b, (short)0, c, false, false);
    // D layout: VGPR r -> row = r + hi*8, col = m
    float* outp = h1 + (row0 + hi * 8) * 64 + (ct << 4) + m;
#pragma unroll
    for (int r = 0; r < 8; ++r) outp[r * 64] = c[r];
  }
}

// layer-1 edge aggregation: agg[dst] += h1[src] * dinv[src]*dinv[dst]
// 16 lanes per edge, float4 per lane -> coalesced 256B gathers, f32 atomics into L2
__global__ void k_agg1(const int* __restrict__ src, const int* __restrict__ dst,
                       const float* __restrict__ dinv, const float* __restrict__ h1,
                       float* __restrict__ agg) {
  unsigned gid = blockIdx.x * blockDim.x + threadIdx.x;  // exactly E*16 threads
  int e = gid >> 4;
  int k = (gid & 15) << 2;
  int s = src[e], d = dst[e];
  float coef = dinv[s] * dinv[d];
  const float4 v = *(const float4*)(h1 + s * 64 + k);
  float* ap = agg + d * 64 + k;
  atomicAdd(ap + 0, v.x * coef);
  atomicAdd(ap + 1, v.y * coef);
  atomicAdd(ap + 2, v.z * coef);
  atomicAdd(ap + 3, v.w * coef);
}

// finalize layer1: h = relu(agg + h1*(2/deg) + b1) in place into agg;
// also accumulate per-column BN sums (LDS reduce -> few global atomics)
__global__ void k_fin1(const float* __restrict__ h1, float* __restrict__ hact,
                       const float* __restrict__ dinv, const float* __restrict__ b1,
                       float* __restrict__ bnsum, float* __restrict__ bnsumsq) {
  int k  = threadIdx.x;   // 0..63 (column)
  int ty = threadIdx.y;   // 0..3
  float bk = b1[k];
  float s = 0.f, ss = 0.f;
  for (int n = blockIdx.x * 4 + ty; n < N_GENES; n += gridDim.x * 4) {
    float di = dinv[n];
    float sc = 2.f * di * di;                  // 2/deg
    float v = hact[n * 64 + k] + h1[n * 64 + k] * sc + bk;
    v = fmaxf(v, 0.f);
    hact[n * 64 + k] = v;
    s += v; ss += v * v;
  }
  __shared__ float ls[4][64], lq[4][64];
  ls[ty][k] = s; lq[ty][k] = ss;
  __syncthreads();
  if (ty == 0) {
    float t  = ls[0][k] + ls[1][k] + ls[2][k] + ls[3][k];
    float t2 = lq[0][k] + lq[1][k] + lq[2][k] + lq[3][k];
    atomicAdd(&bnsum[k], t);
    atomicAdd(&bnsumsq[k], t2);
  }
}

// fold BN (biased var) into W2: h_bn @ W2 = h @ (a*W2) + const
__global__ void k_bnfold(const float* __restrict__ bnsum, const float* __restrict__ bnsumsq,
                         const float* __restrict__ gamma, const float* __restrict__ beta,
                         const float* __restrict__ W2, float* __restrict__ W2f,
                         float* __restrict__ cst) {
  int k = threadIdx.x;   // 64 threads
  float mean = bnsum[k] * (1.f / N_GENES);
  float var  = bnsumsq[k] * (1.f / N_GENES) - mean * mean;
  float a = rsqrtf(var + EPSV) * gamma[k];
  float c = beta[k] - mean * a;
  float w0 = W2[k * 2], w1 = W2[k * 2 + 1];
  W2f[k * 2]     = a * w0;
  W2f[k * 2 + 1] = a * w1;
  __shared__ float r0[64], r1[64];
  r0[k] = c * w0; r1[k] = c * w1;
  __syncthreads();
  if (k == 0) {
    float s0 = 0.f, s1 = 0.f;
    for (int i = 0; i < 64; ++i) { s0 += r0[i]; s1 += r1[i]; }
    cst[0] = s0; cst[1] = s1;
  }
}

// h2 = h_act @ W2f + cst   ([N,64] @ [64,2], W2f staged in LDS)
__global__ void k_gemm2(const float* __restrict__ hact, const float* __restrict__ W2f,
                        const float* __restrict__ cst, float* __restrict__ h2) {
  __shared__ float w[128];
  __shared__ float cs[2];
  if (threadIdx.x < 128) w[threadIdx.x] = W2f[threadIdx.x];
  if (threadIdx.x < 2)   cs[threadIdx.x] = cst[threadIdx.x];
  __syncthreads();
  int n = blockIdx.x * blockDim.x + threadIdx.x;
  if (n >= N_GENES) return;
  float a0 = cs[0], a1 = cs[1];
  const float* hr = hact + n * 64;
#pragma unroll 8
  for (int k = 0; k < 64; ++k) { float hv = hr[k]; a0 += hv * w[2 * k]; a1 += hv * w[2 * k + 1]; }
  h2[n * 2] = a0; h2[n * 2 + 1] = a1;
}

// layer-2 edge aggregation (2 floats per edge)
__global__ void k_agg2(const int* __restrict__ src, const int* __restrict__ dst,
                       const float* __restrict__ dinv, const float* __restrict__ h2,
                       float* __restrict__ agg2) {
  int e = blockIdx.x * blockDim.x + threadIdx.x;   // exactly E threads
  int s = src[e], d = dst[e];
  float coef = dinv[s] * dinv[d];
  float2 v = *(const float2*)(h2 + s * 2);
  atomicAdd(&agg2[d * 2],     v.x * coef);
  atomicAdd(&agg2[d * 2 + 1], v.y * coef);
}

// logits -> log_softmax -> gated accumulate into out
__global__ void k_fin2(const float* __restrict__ agg2, const float* __restrict__ h2,
                       const float* __restrict__ dinv, const float* __restrict__ b2,
                       const float* __restrict__ wgate, int expert, float* __restrict__ out) {
  int n = blockIdx.x * blockDim.x + threadIdx.x;
  if (n >= N_GENES) return;
  float di = dinv[n];
  float sc = 2.f * di * di;
  float l0 = agg2[n * 2]     + h2[n * 2] * sc     + b2[0];
  float l1 = agg2[n * 2 + 1] + h2[n * 2 + 1] * sc + b2[1];
  float m = fmaxf(l0, l1);
  float lse = m + logf(expf(l0 - m) + expf(l1 - m));
  float w = wgate[n * NEXPD + expert];
  out[n * 2]     += w * (l0 - lse);
  out[n * 2 + 1] += w * (l1 - lse);
}

extern "C" void kernel_launch(void* const* d_in, const int* in_sizes, int n_in,
                              void* d_out, int out_size, void* d_ws, size_t ws_size,
                              hipStream_t stream) {
  const float* x     = (const float*)d_in[0];
  const float* gf    = (const float*)d_in[1];
  const float* W1    = (const float*)d_in[2];
  const float* b1    = (const float*)d_in[3];
  const float* gamma = (const float*)d_in[4];
  const float* beta  = (const float*)d_in[5];
  const float* W2    = (const float*)d_in[6];
  const float* b2    = (const float*)d_in[7];
  const float* Wg    = (const float*)d_in[8];
  const float* bg    = (const float*)d_in[9];
  const int*   esrc  = (const int*)d_in[10];
  const int*   edst  = (const int*)d_in[11];
  float* out = (float*)d_out;

  // workspace layout (floats); total ~ N*157 + 2176 floats ~= 63 MB
  float*    W     = (float*)d_ws;
  float*    wgate = W;                                        // N*16
  _Float16* xh    = (_Float16*)(W + (size_t)N_GENES * 16);    // N*16 halves
  float*    h1    = W + (size_t)N_GENES * 24;                 // N*64
  float*    h2    = W + (size_t)N_GENES * 88;                 // N*2
  _Float16* Bfrag = (_Float16*)(W + (size_t)N_GENES * 90);    // 4*32*16 halves (32B-aligned)
  float*    W2f   = W + (size_t)N_GENES * 90 + 1024;          // 128
  float*    cst   = W2f + 128;                                // 2
  float*    Z     = W + (size_t)N_GENES * 90 + 2048;          // per-expert zeroed region
  float*    degv  = Z;                                        // N   (deg -> dinv)
  float*    agg1  = Z + N_GENES;                              // N*64 (then h_act)
  float*    agg2  = Z + (size_t)N_GENES * 65;                 // N*2
  float*    bnsum = Z + (size_t)N_GENES * 67;                 // 64
  float*    bnsq  = bnsum + 64;                               // 64
  const int ZLEN  = N_GENES * 67 + 128;

  k_zero <<<(N_GENES * 2 + 255) / 256, 256, 0, stream>>>(out, N_GENES * 2);
  k_gate <<<(N_GENES + 255) / 256,     256, 0, stream>>>(gf, Wg, bg, wgate);
  k_xconv<<<(N_GENES * 16) / 256,      256, 0, stream>>>(x, xh);

  for (int i = 0; i < NEXPD; ++i) {
    const int* si = esrc + (size_t)i * E_EDGES;
    const int* di = edst + (size_t)i * E_EDGES;
    k_zero <<<(ZLEN + 255) / 256, 256, 0, stream>>>(Z, ZLEN);
    k_bswz <<<2048 / 256, 256, 0, stream>>>(W1 + (size_t)i * UNITD * H1D, Bfrag);
    k_deg  <<<E_EDGES / 256, 256, 0, stream>>>(di, degv);
    k_dinv <<<(N_GENES + 255) / 256, 256, 0, stream>>>(degv);
    k_gemm1<<<3125, 64, 0, stream>>>(xh, Bfrag, h1);
    k_agg1 <<<(E_EDGES * 16) / 256, 256, 0, stream>>>(si, di, degv, h1, agg1);
    dim3 b14(64, 4);
    k_fin1 <<<512, b14, 0, stream>>>(h1, agg1, degv, b1 + i * H1D, bnsum, bnsq);
    k_bnfold<<<1, 64, 0, stream>>>(bnsum, bnsq, gamma + i * H1D, beta + i * H1D,
                                   W2 + i * H1D * 2, W2f, cst);
    k_gemm2<<<(N_GENES + 255) / 256, 256, 0, stream>>>(agg1, W2f, cst, h2);
    k_agg2 <<<E_EDGES / 256, 256, 0, stream>>>(si, di, degv, h2, agg2);
    k_fin2 <<<(N_GENES + 255) / 256, 256, 0, stream>>>(agg2, h2, degv, b2 + i * 2,
                                                       wgate, i, out);
  }
  (void)in_sizes; (void)n_in; (void)out_size; (void)ws_size;
}
